// TFTransfoXLMainLayer_41497974014449
// MI455X (gfx1250) — matrix operand order repaired and checked by reading, not compiled, not run on hardware
//
#include <hip/hip_runtime.h>
#include <hip/hip_bf16.h>

typedef __attribute__((ext_vector_type(16))) __bf16 v16bf;
typedef __attribute__((ext_vector_type(8)))  float  v8f;

union Frag { v16bf v; uint4 q[2]; unsigned short u[16]; };

__device__ __forceinline__ unsigned short f2bf(float f) {
    unsigned int u = __builtin_bit_cast(unsigned int, f);
    unsigned int r = u + 0x7fffu + ((u >> 16) & 1u);   // round-to-nearest-even
    return (unsigned short)(r >> 16);
}
__device__ __forceinline__ float bf2f(unsigned short h) {
    unsigned int u = ((unsigned int)h) << 16;
    return __builtin_bit_cast(float, u);
}

// ---------------------------------------------------------------------------
// Tiled fp32 [K,N] -> bf16 transposed [N,K]; blockIdx.z = layer.
// ---------------------------------------------------------------------------
__global__ __launch_bounds__(256) void convert_transpose_kernel(
    const float* __restrict__ in, unsigned short* __restrict__ out,
    int K, int N) {
    __shared__ float tile[32][33];
    long base = (long)blockIdx.z * K * N;
    int k0 = blockIdx.y * 32, n0 = blockIdx.x * 32;
    int tx = threadIdx.x & 31, ty = threadIdx.x >> 5;
    for (int i = ty; i < 32; i += 8)
        tile[i][tx] = in[base + (long)(k0 + i) * N + n0 + tx];
    __syncthreads();
    for (int i = ty; i < 32; i += 8)
        out[base + (long)(n0 + i) * K + k0 + tx] = f2bf(tile[tx][i]);
}

// ---------------------------------------------------------------------------
// Sinusoidal relative positional embedding, bf16 out. [klen=1024, 1024]
// ---------------------------------------------------------------------------
__global__ void pos_emb_kernel(unsigned short* __restrict__ pos_bf) {
    int idx = blockIdx.x * blockDim.x + threadIdx.x;
    if (idx >= 1024 * 1024) return;
    int r = idx >> 10, c = idx & 1023;
    float pos = (float)(1023 - r);
    int k = (c < 512) ? c : (c - 512);
    float inv = __expf(-((float)(2 * k) * (1.0f / 1024.0f)) * 9.210340371976184f);
    float ang = pos * inv;
    float val = (c < 512) ? __sinf(ang) : __cosf(ang);
    pos_bf[idx] = f2bf(val);
}

// ---------------------------------------------------------------------------
// Adaptive embedding: one block per token, gather + matvec projection.
// ---------------------------------------------------------------------------
__global__ __launch_bounds__(256) void embed_kernel(
    const int* __restrict__ ids,
    const float* __restrict__ emb0, const float* __restrict__ proj0,
    const float* __restrict__ emb1, const float* __restrict__ proj1,
    const float* __restrict__ emb2, const float* __restrict__ proj2,
    float* __restrict__ h) {
    int token = blockIdx.x;            // q*8 + b
    int q = token >> 3, b = token & 7;
    int id = ids[b * 512 + q];
    const float* tab; const float* proj; int d_i; int lo;
    if (id < 2000)       { tab = emb0; proj = proj0; d_i = 1024; lo = 0; }
    else if (id < 10000) { tab = emb1; proj = proj1; d_i = 256;  lo = 2000; }
    else                 { tab = emb2; proj = proj2; d_i = 64;   lo = 10000; }
    __shared__ float e[1024];
    const float* row = tab + (long)(id - lo) * d_i;
    for (int t = threadIdx.x; t < d_i; t += 256) e[t] = row[t];
    __syncthreads();
    for (int d = threadIdx.x; d < 1024; d += 256) {
        float s = 0.f;
        for (int k = 0; k < d_i; ++k) s += e[k] * proj[(long)k * 1024 + d];
        h[(long)token * 1024 + d] = s * 32.0f;   // * sqrt(d_model)
    }
}

// ---------------------------------------------------------------------------
// cat_bf [klen=1024, 8, 1024] = bf16([mems_l ; h])
// ---------------------------------------------------------------------------
__global__ void build_cat_kernel(const float* __restrict__ mems_l,
                                 const float* __restrict__ h,
                                 unsigned short* __restrict__ cat_bf) {
    long idx = (long)blockIdx.x * blockDim.x + threadIdx.x;
    if (idx >= (long)1024 * 8 * 1024) return;
    long j = idx >> 13;
    float v = (j < 512) ? mems_l[idx] : h[idx - (long)512 * 8192];
    cat_bf[idx] = f2bf(v);
}

// ---------------------------------------------------------------------------
// Batched strided bf16 WMMA GEMM: for batch z (b = z/16, n = z%16):
//   C[z] = A[z] (MxK, row ld lda) @ Bt[z] (NxK, row ld ldb)^T
// Block tile 128x64, 8 waves, each wave 2x2 WMMA tiles.
// Requires M%128==0, N%64==0, K%32==0.
// ---------------------------------------------------------------------------
__global__ __launch_bounds__(256) void gemm_bf16_wmma(
    const unsigned short* __restrict__ A,  long sAb, long sAn, int lda,
    const unsigned short* __restrict__ Bt, long sBb, long sBn, int ldb,
    const float* __restrict__ bias,
    float* __restrict__ Cf, unsigned short* __restrict__ Cb,
    long sCb, long sCn, int ldc,
    int M, int N, int K, int relu) {
    __shared__ __align__(16) unsigned short As[8 * 32 * 16];   // fragment order
    __shared__ __align__(16) unsigned short Bs[64 * 32];       // row-major
    const int tid  = threadIdx.x;
    const int lane = tid & 31;
    const int wave = tid >> 5;
    const int wm = wave & 3;
    const int wn = wave >> 2;
    const int bm = blockIdx.y * 128;
    const int bn = blockIdx.x * 64;
    const int n15  = lane & 15;
    const int half = lane >> 4;
    const long z  = blockIdx.z;
    const long zb = z >> 4, zn = z & 15;
    const unsigned short* Az  = A  + zb * sAb + zn * sAn;
    const unsigned short* Btz = Bt + zb * sBb + zn * sBn;

    v8f acc[2][2] = {};

    for (int k0 = 0; k0 < K; k0 += 32) {
        #pragma unroll
        for (int it = 0; it < 2; ++it) {
            int linear = tid + it * 256;
            int arow = linear >> 2;
            int c    = linear & 3;
            uint4 val = *(const uint4*)(Az + (long)(bm + arow) * lda + k0 + c * 8);
            int dst = (((arow >> 4) * 32 + (c & 1) * 16 + (arow & 15)) << 4)
                      + (c >> 1) * 8;
            *(uint4*)(&As[dst]) = val;
        }
        {
            int brow = tid >> 2, c = tid & 3;
            uint4 val = *(const uint4*)(Btz + (long)(bn + brow) * ldb + k0 + c * 8);
            *(uint4*)(&Bs[brow * 32 + c * 8]) = val;
        }
        if (k0 + 32 < K) {
            __builtin_prefetch(Az + (long)(bm + (tid >> 2)) * lda + k0 + 32, 0, 3);
            __builtin_prefetch(Btz + (long)(bn + (tid >> 2)) * ldb + k0 + 32, 0, 3);
        }
        __syncthreads();

        Frag af[2], bfr[2];
        #pragma unroll
        for (int i = 0; i < 2; ++i) {
            const unsigned short* p = &As[((wm * 2 + i) * 32 + lane) << 4];
            af[i].q[0] = *(const uint4*)(p);
            af[i].q[1] = *(const uint4*)(p + 8);
        }
        #pragma unroll
        for (int j = 0; j < 2; ++j) {
            const unsigned short* p = &Bs[(wn * 32 + j * 16 + n15) * 32 + half * 8];
            bfr[j].q[0] = *(const uint4*)(p);
            bfr[j].q[1] = *(const uint4*)(p + 16);
        }
        #pragma unroll
        for (int i = 0; i < 2; ++i)
            #pragma unroll
            for (int j = 0; j < 2; ++j)
                acc[i][j] = __builtin_amdgcn_wmma_f32_16x16x32_bf16(
                    false, af[i].v, false, bfr[j].v, (short)0, acc[i][j],
                    false, false);
        __syncthreads();
    }

    #pragma unroll
    for (int i = 0; i < 2; ++i) {
        #pragma unroll
        for (int j = 0; j < 2; ++j) {
            int col = bn + wn * 32 + j * 16 + n15;
            float bv = bias ? bias[col] : 0.f;
            #pragma unroll
            for (int v = 0; v < 8; ++v) {
                int row = bm + wm * 32 + i * 16 + half * 8 + v;
                float c0 = acc[i][j][v] + bv;
                if (relu) c0 = fmaxf(c0, 0.f);
                long cidx = zb * sCb + zn * sCn + (long)row * ldc + col;
                if (Cf) Cf[cidx] = c0;
                if (Cb) Cb[cidx] = f2bf(c0);
            }
        }
    }
}

// ---------------------------------------------------------------------------
// qw/qr [z=b*16+n][512][64] = bf16(q + r_w_bias), bf16(q + r_r_bias)
// ---------------------------------------------------------------------------
__global__ void build_q_kernel(const unsigned short* __restrict__ heads,
                               const float* __restrict__ rwb,
                               const float* __restrict__ rrb,
                               unsigned short* __restrict__ qw,
                               unsigned short* __restrict__ qr) {
    long idx = (long)blockIdx.x * blockDim.x + threadIdx.x;
    if (idx >= (long)128 * 512 * 64) return;
    int d = (int)(idx & 63);
    int i = (int)((idx >> 6) & 511);
    long z = idx >> 15;
    int n = (int)(z & 15), b = (int)(z >> 4);
    float q = bf2f(heads[((long)(512 + i) * 8 + b) * 3072 + n * 64 + d]);
    qw[idx] = f2bf(q + rwb[n * 64 + d]);
    qr[idx] = f2bf(q + rrb[n * 64 + d]);
}

// ---------------------------------------------------------------------------
// Vt [z][64 d][1024 j] = V[j][d] for batch z=b*16+n (tiled transpose)
// ---------------------------------------------------------------------------
__global__ __launch_bounds__(256) void v_transpose_kernel(
    const unsigned short* __restrict__ heads, unsigned short* __restrict__ Vt) {
    __shared__ unsigned short t[32][33];
    long z = blockIdx.z;
    int b = (int)(z & 7), n = (int)(z >> 3);           // z = n*8 + b? keep z=b*16+n:
    n = (int)(z & 15); b = (int)(z >> 4);
    int j0 = blockIdx.x * 32, d0 = blockIdx.y * 32;
    int tx = threadIdx.x & 31, ty = threadIdx.x >> 5;
    for (int i = ty; i < 32; i += 8)
        t[i][tx] = heads[((long)(j0 + i) * 8 + b) * 3072 + 2048 + n * 64 + d0 + tx];
    __syncthreads();
    for (int i = ty; i < 32; i += 8)
        Vt[((long)z * 64 + d0 + i) * 1024 + j0 + tx] = t[tx][i];
}

// ---------------------------------------------------------------------------
// Softmax with rel-shift: row (z, i). P = softmax((AC + BD_shift) * scale),
// written in place over AC (bf16). BD_shift[i,j] = BD[i, j - i + 511].
// ---------------------------------------------------------------------------
__global__ __launch_bounds__(256) void softmax_rel_kernel(
    unsigned short* __restrict__ AC,       // [128][512][1024]: in AC, out P
    const unsigned short* __restrict__ BD) // [128][512][1024]
{
    const int i = blockIdx.x;
    const long z = blockIdx.y;
    unsigned short* acr = AC + (z * 512 + i) * 1024;
    const unsigned short* bdr = BD + (z * 512 + i) * 1024;
    __shared__ float sc[1024];
    __shared__ float red[256];
    const int tid = threadIdx.x;

    for (int jj = 0; jj < 4; ++jj) {
        int j = tid + jj * 256;
        float s = -1e30f;
        if (j <= i + 512)
            s = (bf2f(acr[j]) + bf2f(bdr[j - i + 511])) * 0.125f;
        sc[j] = s;
    }
    __syncthreads();
    float mx = -1e30f;
    for (int jj = 0; jj < 4; ++jj) mx = fmaxf(mx, sc[tid + jj * 256]);
    red[tid] = mx; __syncthreads();
    for (int s = 128; s > 0; s >>= 1) {
        if (tid < s) red[tid] = fmaxf(red[tid], red[tid + s]);
        __syncthreads();
    }
    mx = red[0]; __syncthreads();
    float lsum = 0.f;
    for (int jj = 0; jj < 4; ++jj) {
        int j = tid + jj * 256;
        float p = __expf(sc[j] - mx);
        sc[j] = p;
        lsum += p;
    }
    red[tid] = lsum; __syncthreads();
    for (int s = 128; s > 0; s >>= 1) {
        if (tid < s) red[tid] += red[tid + s];
        __syncthreads();
    }
    float inv = 1.0f / red[0];
    for (int jj = 0; jj < 4; ++jj) {
        int j = tid + jj * 256;
        acr[j] = (j <= i + 512) ? f2bf(sc[j] * inv) : (unsigned short)0;
    }
}

// ---------------------------------------------------------------------------
// vec[(i*8+b)*1024 + n*64 + d] = pv[((b*16+n)*512 + i)*64 + d]
// ---------------------------------------------------------------------------
__global__ void pv_reorder_kernel(const unsigned short* __restrict__ pv,
                                  unsigned short* __restrict__ vec) {
    long idx = (long)blockIdx.x * blockDim.x + threadIdx.x;
    if (idx >= (long)128 * 512 * 64) return;
    int d = (int)(idx & 63);
    int i = (int)((idx >> 6) & 511);
    long z = idx >> 15;
    int n = (int)(z & 15), b = (int)(z >> 4);
    vec[((long)i * 8 + b) * 1024 + n * 64 + d] = pv[idx];
}

// ---------------------------------------------------------------------------
// h = LayerNorm(h + delta) * g + b ; also bf16 copy of result.
// ---------------------------------------------------------------------------
__global__ __launch_bounds__(256) void add_ln_kernel(
    float* __restrict__ h, const float* __restrict__ delta,
    const float* __restrict__ g, const float* __restrict__ bb,
    unsigned short* __restrict__ h_bf) {
    const int row = blockIdx.x;
    const int tid = threadIdx.x;
    __shared__ float red[256];
    float x[4];
    float lsum = 0.f;
    for (int t = 0; t < 4; ++t) {
        int d = tid + t * 256;
        x[t] = h[(long)row * 1024 + d] + delta[(long)row * 1024 + d];
        lsum += x[t];
    }
    red[tid] = lsum; __syncthreads();
    for (int s = 128; s > 0; s >>= 1) {
        if (tid < s) red[tid] += red[tid + s];
        __syncthreads();
    }
    float mu = red[0] * (1.0f / 1024.0f); __syncthreads();
    float lv = 0.f;
    for (int t = 0; t < 4; ++t) { float dv = x[t] - mu; lv += dv * dv; }
    red[tid] = lv; __syncthreads();
    for (int s = 128; s > 0; s >>= 1) {
        if (tid < s) red[tid] += red[tid + s];
        __syncthreads();
    }
    float rstd = rsqrtf(red[0] * (1.0f / 1024.0f) + 1e-5f);
    for (int t = 0; t < 4; ++t) {
        int d = tid + t * 256;
        float y = (x[t] - mu) * rstd * g[d] + bb[d];
        h[(long)row * 1024 + d] = y;
        h_bf[(long)row * 1024 + d] = f2bf(y);
    }
}

// ---------------------------------------------------------------------------
// out[b,q,d] = h[q,b,d]
// ---------------------------------------------------------------------------
__global__ void out_kernel(const float* __restrict__ h, float* __restrict__ out) {
    long idx = (long)blockIdx.x * blockDim.x + threadIdx.x;
    if (idx >= (long)512 * 8 * 1024) return;
    int d = (int)(idx & 1023);
    long rest = idx >> 10;
    int b = (int)(rest & 7);
    long q = rest >> 3;
    out[((long)b * 512 + q) * 1024 + d] = h[idx];
}

// ---------------------------------------------------------------------------
extern "C" void kernel_launch(void* const* d_in, const int* in_sizes, int n_in,
                              void* d_out, int out_size, void* d_ws, size_t ws_size,
                              hipStream_t stream) {
    (void)in_sizes; (void)n_in; (void)out_size; (void)ws_size;
    const int*   ids    = (const int*)  d_in[0];
    const float* mems   = (const float*)d_in[1];
    const float* emb0   = (const float*)d_in[2];
    const float* proj0  = (const float*)d_in[3];
    const float* emb1   = (const float*)d_in[4];
    const float* proj1  = (const float*)d_in[5];
    const float* emb2   = (const float*)d_in[6];
    const float* proj2  = (const float*)d_in[7];
    const float* qkv_w  = (const float*)d_in[8];
    const float* r_w    = (const float*)d_in[9];
    const float* o_w    = (const float*)d_in[10];
    const float* ln1_g  = (const float*)d_in[11];
    const float* ln1_b  = (const float*)d_in[12];
    const float* ff_w1  = (const float*)d_in[13];
    const float* ff_b1  = (const float*)d_in[14];
    const float* ff_w2  = (const float*)d_in[15];
    const float* ff_b2  = (const float*)d_in[16];
    const float* ln2_g  = (const float*)d_in[17];
    const float* ln2_b  = (const float*)d_in[18];
    const float* rwbias = (const float*)d_in[19];
    const float* rrbias = (const float*)d_in[20];

    char* ws = (char*)d_ws;
    size_t off = 0;
    auto carve = [&](size_t bytes) {
        void* p = ws + off;
        off = (off + bytes + 255) & ~(size_t)255;
        return p;
    };
    const long NQKV = 6L * 1024 * 3072;
    const long NRW  = 6L * 1024 * 1024;
    const long NFF1 = 6L * 1024 * 4096;
    unsigned short* qkvT_bf = (unsigned short*)carve(NQKV * 2);
    unsigned short* rwT_bf  = (unsigned short*)carve(NRW * 2);
    unsigned short* owT_bf  = (unsigned short*)carve(NRW * 2);
    unsigned short* ff1T_bf = (unsigned short*)carve(NFF1 * 2);
    unsigned short* ff2T_bf = (unsigned short*)carve(NFF1 * 2);
    unsigned short* pos_bf  = (unsigned short*)carve(1024L * 1024 * 2);
    float*          h       = (float*)carve(4096L * 1024 * 4);
    unsigned short* h_bf    = (unsigned short*)carve(4096L * 1024 * 2);
    unsigned short* cat_bf  = (unsigned short*)carve(8192L * 1024 * 2);
    unsigned short* heads_bf= (unsigned short*)carve(8192L * 3072 * 2);
    unsigned short* rk_bf   = (unsigned short*)carve(1024L * 1024 * 2);
    unsigned short* vec_bf  = (unsigned short*)carve(4096L * 1024 * 2);
    float*          tmp     = (float*)carve(4096L * 1024 * 4);
    unsigned short* ffact_bf= (unsigned short*)carve(4096L * 4096 * 2);
    unsigned short* qw_bf   = (unsigned short*)carve(128L * 512 * 64 * 2);
    unsigned short* qr_bf   = (unsigned short*)carve(128L * 512 * 64 * 2);
    unsigned short* AC_bf   = (unsigned short*)carve(128L * 512 * 1024 * 2);
    unsigned short* BD_bf   = (unsigned short*)carve(128L * 512 * 1024 * 2);
    unsigned short* Vt_bf   = (unsigned short*)carve(128L * 64 * 1024 * 2);
    unsigned short* pv_bf   = (unsigned short*)carve(128L * 512 * 64 * 2);

    convert_transpose_kernel<<<dim3(96, 32, 6), 256, 0, stream>>>(qkv_w, qkvT_bf, 1024, 3072);
    convert_transpose_kernel<<<dim3(32, 32, 6), 256, 0, stream>>>(r_w,  rwT_bf, 1024, 1024);
    convert_transpose_kernel<<<dim3(32, 32, 6), 256, 0, stream>>>(o_w,  owT_bf, 1024, 1024);
    convert_transpose_kernel<<<dim3(128, 32, 6), 256, 0, stream>>>(ff_w1, ff1T_bf, 1024, 4096);
    convert_transpose_kernel<<<dim3(32, 128, 6), 256, 0, stream>>>(ff_w2, ff2T_bf, 4096, 1024);

    pos_emb_kernel<<<4096, 256, 0, stream>>>(pos_bf);
    embed_kernel<<<4096, 256, 0, stream>>>(ids, emb0, proj0, emb1, proj1,
                                           emb2, proj2, h);

    for (int l = 0; l < 6; ++l) {
        build_cat_kernel<<<(8192 * 1024) / 256, 256, 0, stream>>>(
            mems + (long)l * 512 * 8 * 1024, h, cat_bf);

        // heads = cat @ qkv_w[l] : [8192 x 3072], K=1024
        gemm_bf16_wmma<<<dim3(48, 64, 1), 256, 0, stream>>>(
            cat_bf, 0, 0, 1024,
            qkvT_bf + (long)l * 1024 * 3072, 0, 0, 1024,
            nullptr, nullptr, heads_bf, 0, 0, 3072,
            8192, 3072, 1024, 0);

        // rk = pos_emb @ r_w[l] : [1024 x 1024]
        gemm_bf16_wmma<<<dim3(16, 8, 1), 256, 0, stream>>>(
            pos_bf, 0, 0, 1024,
            rwT_bf + (long)l * 1024 * 1024, 0, 0, 1024,
            nullptr, nullptr, rk_bf, 0, 0, 1024,
            1024, 1024, 1024, 0);

        build_q_kernel<<<(128 * 512 * 64) / 256, 256, 0, stream>>>(
            heads_bf, rwbias + l * 1024, rrbias + l * 1024, qw_bf, qr_bf);
        v_transpose_kernel<<<dim3(32, 2, 128), 256, 0, stream>>>(heads_bf, Vt_bf);

        // AC[z] = qw[z] @ K[z]^T : M=512, N=1024(klen), K=64, 128 batches
        gemm_bf16_wmma<<<dim3(16, 4, 128), 256, 0, stream>>>(
            qw_bf, 16L * 512 * 64, 512L * 64, 64,
            heads_bf + 1024, 3072, 64, 8 * 3072,
            nullptr, nullptr, AC_bf, 16L * 512 * 1024, 512L * 1024, 1024,
            512, 1024, 64, 0);

        // BD_raw[z] = qr[z] @ rk[n]^T : same shape; rk has no batch-b stride
        gemm_bf16_wmma<<<dim3(16, 4, 128), 256, 0, stream>>>(
            qr_bf, 16L * 512 * 64, 512L * 64, 64,
            rk_bf, 0, 64, 1024,
            nullptr, nullptr, BD_bf, 16L * 512 * 1024, 512L * 1024, 1024,
            512, 1024, 64, 0);

        softmax_rel_kernel<<<dim3(512, 128), 256, 0, stream>>>(AC_bf, BD_bf);

        // pv[z] = P[z] @ V[z] : M=512, N=64, K=1024
        gemm_bf16_wmma<<<dim3(1, 4, 128), 256, 0, stream>>>(
            AC_bf, 16L * 512 * 1024, 512L * 1024, 1024,
            Vt_bf, 16L * 64 * 1024, 64L * 1024, 1024,
            nullptr, nullptr, pv_bf, 16L * 512 * 64, 512L * 64, 64,
            512, 64, 1024, 0);

        pv_reorder_kernel<<<(128 * 512 * 64) / 256, 256, 0, stream>>>(pv_bf, vec_bf);

        // tmp = vec @ o_w[l] : [4096 x 1024]
        gemm_bf16_wmma<<<dim3(16, 32, 1), 256, 0, stream>>>(
            vec_bf, 0, 0, 1024,
            owT_bf + (long)l * 1024 * 1024, 0, 0, 1024,
            nullptr, tmp, nullptr, 0, 0, 1024,
            4096, 1024, 1024, 0);

        add_ln_kernel<<<4096, 256, 0, stream>>>(
            h, tmp, ln1_g + l * 1024, ln1_b + l * 1024, h_bf);

        // ffact = relu(h @ ff_w1[l] + b1) : [4096 x 4096]
        gemm_bf16_wmma<<<dim3(64, 32, 1), 256, 0, stream>>>(
            h_bf, 0, 0, 1024,
            ff1T_bf + (long)l * 1024 * 4096, 0, 0, 1024,
            ff_b1 + l * 4096, nullptr, ffact_bf, 0, 0, 4096,
            4096, 4096, 1024, 1);

        // tmp = ffact @ ff_w2[l] + b2 : [4096 x 1024], K=4096
        gemm_bf16_wmma<<<dim3(16, 32, 1), 256, 0, stream>>>(
            ffact_bf, 0, 0, 4096,
            ff2T_bf + (long)l * 4096 * 1024, 0, 0, 4096,
            ff_b2 + l * 1024, tmp, nullptr, 0, 0, 1024,
            4096, 1024, 4096, 0);

        add_ln_kernel<<<4096, 256, 0, stream>>>(
            h, tmp, ln2_g + l * 1024, ln2_b + l * 1024, h_bf);
    }

    out_kernel<<<(4096 * 1024) / 256, 256, 0, stream>>>(h, (float*)d_out);
}
